// MPNN_76536317215339
// MI455X (gfx1250) — compile-verified, matrix operand order checked by reading, not run on hardware
//
#include <hip/hip_runtime.h>

#define NN 32768
#define EE 262144
#define BB 512
#define HH 128
#define LLAYERS 3
#define WPB 4   // waves per block (wave32)

typedef __attribute__((ext_vector_type(16))) __bf16 v16bf;
typedef __attribute__((ext_vector_type(8)))  float  v8f;

union FragBF { unsigned u[8]; v16bf v; };

// ---------------------------------------------------------------- helpers ---

__device__ __forceinline__ unsigned pk_bf16(float a, float b) {
  unsigned ua = __float_as_uint(a), ub = __float_as_uint(b);
  ua = (ua + 0x7FFFu + ((ua >> 16) & 1u)) >> 16;          // rne -> low half
  ub = (ub + 0x7FFFu + ((ub >> 16) & 1u)) & 0xFFFF0000u;  // rne -> high half
  return ua | ub;
}

// 16x32 bf16 A/B fragment from a bf16 row (viewed as dwords, 2 elems each).
// Lane half 0: dwords {0..3, 8..11}  (K = 0..7, 16..23)
// Lane half 1: dwords {4..7, 12..15} (K = 8..15, 24..31)
__device__ __forceinline__ v16bf frag_bf16(const unsigned* p) {
  FragBF f;
  const int h = (threadIdx.x & 16) >> 2;   // 0 or 4 dwords
#pragma unroll
  for (int j = 0; j < 4; ++j) f.u[j]     = p[h + j];
#pragma unroll
  for (int j = 0; j < 4; ++j) f.u[4 + j] = p[8 + h + j];
  return f.v;
}

// Same fragment, but sourced from f32 data converted on the fly.
__device__ __forceinline__ v16bf frag_f32(const float* p) {
  FragBF f;
  const int h = (threadIdx.x & 16) >> 1;   // 0 or 8 elements
#pragma unroll
  for (int j = 0; j < 4; ++j) { int e = h + 2 * j;      f.u[j]     = pk_bf16(p[e], p[e + 1]); }
#pragma unroll
  for (int j = 0; j < 4; ++j) { int e = 16 + h + 2 * j; f.u[4 + j] = pk_bf16(p[e], p[e + 1]); }
  return f.v;
}

__device__ __forceinline__ v8f wmma_bf16(v16bf a, v16bf b, v8f c) {
  return __builtin_amdgcn_wmma_f32_16x16x32_bf16(false, a, false, b, (short)0, c,
                                                 false, false);
}

// ------------------------------------------------------------ tiny kernels ---

__global__ void zero_kernel(float* p, int n) {
  int i = blockIdx.x * blockDim.x + threadIdx.x;
  if (i < n) p[i] = 0.0f;
}

// pairwise f32 -> packed bf16 (layer-0 input mirrors only)
__global__ void cvt_bf16_kernel(const float* __restrict__ src, unsigned* __restrict__ dst,
                                int ndw) {
  int i = blockIdx.x * blockDim.x + threadIdx.x;
  if (i < ndw) dst[i] = pk_bf16(src[2 * i], src[2 * i + 1]);
}

// weight [K][128] f32 -> transposed bf16 [128][K]
__global__ void wt_cvt_kernel(const float* __restrict__ src, unsigned short* __restrict__ dst,
                              int K) {
  int idx = blockIdx.x * blockDim.x + threadIdx.x;
  if (idx >= 128 * K) return;
  int h = idx / K, k = idx - h * K;
  unsigned uv = __float_as_uint(src[(size_t)k * 128 + h]);
  uv = (uv + 0x7FFFu + ((uv >> 16) & 1u)) >> 16;
  dst[idx] = (unsigned short)uv;
}

// per-column sum / sumsq reduction (stats[0..127]=sum, [128..255]=sumsq)
__global__ void bn_stat_kernel(const float* __restrict__ v, float* __restrict__ stats,
                               int rows) {
  int c = threadIdx.x & 127;
  int r0 = blockIdx.x * 2 + (threadIdx.x >> 7);
  float s = 0.f, s2 = 0.f;
  for (int r = r0; r < rows; r += gridDim.x * 2) {
    float x = v[(size_t)r * HH + c];
    s += x; s2 += x * x;
  }
  atomicAdd(&stats[c], s);
  atomicAdd(&stats[128 + c], s2);
}

// normalize in place AND emit the bf16 mirror for the next layer (fused)
__global__ void bn_norm_kernel(float* __restrict__ v, unsigned* __restrict__ vbf,
                               const float* __restrict__ stats,
                               const float* __restrict__ g, const float* __restrict__ b,
                               int rows) {
  int i = blockIdx.x * blockDim.x + threadIdx.x;   // pair index
  if (i >= rows * (HH / 2)) return;
  int c0 = (2 * i) & 127;
  float inv_n = 1.0f / (float)rows;
  float m0 = stats[c0] * inv_n,     m1 = stats[c0 + 1] * inv_n;
  float s0 = rsqrtf(stats[128 + c0] * inv_n - m0 * m0 + 1e-5f);
  float s1 = rsqrtf(stats[128 + c0 + 1] * inv_n - m1 * m1 + 1e-5f);
  float v0 = (v[2 * i] - m0) * s0 * g[c0] + b[c0];
  float v1 = (v[2 * i + 1] - m1) * s1 * g[c0 + 1] + b[c0 + 1];
  v[2 * i] = v0;
  v[2 * i + 1] = v1;
  vbf[i] = pk_bf16(v0, v1);
}

// ------------------------------------------------------------- edge kernel ---
// One wave per 32-edge tile (two 16-row sub-tiles sharing every B fragment).
// Fused: edge MLP (K=512), edge_agg scatter, message MLP (K=256, second half
// transposed through LDS), agg scatter.
__global__ void __launch_bounds__(WPB * 32)
edge_kernel(const unsigned* __restrict__ xbf, const unsigned* __restrict__ eabf,
            const unsigned* __restrict__ ubf, const int* __restrict__ eidx,
            const int* __restrict__ batch,
            const unsigned* __restrict__ WeT, const float* __restrict__ We_b,
            const unsigned* __restrict__ Wn1T, const float* __restrict__ Wn1_b,
            float* __restrict__ ea_out, float* __restrict__ agg,
            float* __restrict__ edge_agg, int act) {
  __shared__ float smem[WPB][32 * HH];

  const int wave  = threadIdx.x >> 5;
  const int lane  = threadIdx.x & 31;
  const int r     = lane & 15;
  const int half8 = (lane & 16) >> 1;           // C-layout row offset (0 or 8)
  const int e0    = (blockIdx.x * WPB + wave) * 32;

  // per-lane A-row metadata (row r of sub-tile 0 and sub-tile 1)
  const int d0 = eidx[e0 + r],      s0 = eidx[EE + e0 + r];
  const int d1 = eidx[e0 + 16 + r], s1 = eidx[EE + e0 + 16 + r];
  const unsigned* baseA0[4] = { xbf + (size_t)d0 * 64, xbf + (size_t)s0 * 64,
                                eabf + (size_t)(e0 + r) * 64,
                                ubf + (size_t)batch[d0] * 64 };
  const unsigned* baseA1[4] = { xbf + (size_t)d1 * 64, xbf + (size_t)s1 * 64,
                                eabf + (size_t)(e0 + 16 + r) * 64,
                                ubf + (size_t)batch[d1] * 64 };

  __builtin_prefetch(WeT, 0, 1);                // global_prefetch_b8 on weights

  const v8f vzero = {0.f, 0.f, 0.f, 0.f, 0.f, 0.f, 0.f, 0.f};
  v8f acc0[8], acc1[8];
#pragma unroll
  for (int t = 0; t < 8; ++t) { acc0[t] = vzero; acc1[t] = vzero; }

  // --- edge GEMM: D = concat4 @ We  (K = 512, WeT rows are 256 dwords) ---
#pragma unroll
  for (int s = 0; s < 4; ++s) {
#pragma unroll
    for (int kb = 0; kb < 4; ++kb) {
      v16bf a0 = frag_bf16(baseA0[s] + kb * 16);
      v16bf a1 = frag_bf16(baseA1[s] + kb * 16);
      const int kk = s * 64 + kb * 16;
#pragma unroll
      for (int nt = 0; nt < 8; ++nt) {
        v16bf b = frag_bf16(WeT + (size_t)(nt * 16 + r) * 256 + kk);
        acc0[nt] = wmma_bf16(a0, b, acc0[nt]);
        acc1[nt] = wmma_bf16(a1, b, acc1[nt]);
      }
    }
  }

  // epilogue per sub-tile: bias/act, store e, LDS stage, edge_agg scatter
  float* sm = smem[wave];
#pragma unroll
  for (int t = 0; t < 2; ++t) {
    v8f* acc = t ? acc1 : acc0;
    const int eb = e0 + t * 16;
    int ebm[8];
#pragma unroll
    for (int v = 0; v < 8; ++v) ebm[v] = batch[eidx[eb + v + half8]];
#pragma unroll
    for (int nt = 0; nt < 8; ++nt) {
      int n = nt * 16 + r;
      float bia = We_b[n];
#pragma unroll
      for (int v = 0; v < 8; ++v) {
        int m = v + half8;
        float val = acc[nt][v] + bia;
        if (act) val = fmaxf(val, 0.f);
        ea_out[(size_t)(eb + m) * HH + n] = val;
        sm[(t * 16 + m) * HH + n] = val;
        atomicAdd(&edge_agg[(size_t)ebm[v] * HH + n], val);
      }
    }
  }
  __syncthreads();

  // --- message GEMM: m = act([x[col], e] @ Wn1)  (K = 256, rows 128 dwords) ---
  v8f mc0[8], mc1[8];
#pragma unroll
  for (int t = 0; t < 8; ++t) { mc0[t] = vzero; mc1[t] = vzero; }

#pragma unroll
  for (int kb = 0; kb < 4; ++kb) {                       // K-half 0: x[col]
    v16bf a0 = frag_bf16(baseA0[1] + kb * 16);
    v16bf a1 = frag_bf16(baseA1[1] + kb * 16);
#pragma unroll
    for (int nt = 0; nt < 8; ++nt) {
      v16bf b = frag_bf16(Wn1T + (size_t)(nt * 16 + r) * 128 + kb * 16);
      mc0[nt] = wmma_bf16(a0, b, mc0[nt]);
      mc1[nt] = wmma_bf16(a1, b, mc1[nt]);
    }
  }
#pragma unroll
  for (int kb = 0; kb < 4; ++kb) {                       // K-half 1: e (LDS)
    v16bf a0 = frag_f32(sm + r * HH + kb * 32);
    v16bf a1 = frag_f32(sm + (16 + r) * HH + kb * 32);
#pragma unroll
    for (int nt = 0; nt < 8; ++nt) {
      v16bf b = frag_bf16(Wn1T + (size_t)(nt * 16 + r) * 128 + 64 + kb * 16);
      mc0[nt] = wmma_bf16(a0, b, mc0[nt]);
      mc1[nt] = wmma_bf16(a1, b, mc1[nt]);
    }
  }

  // bias + act + scatter-sum into agg[dest]
#pragma unroll
  for (int t = 0; t < 2; ++t) {
    v8f* mc = t ? mc1 : mc0;
    const int eb = e0 + t * 16;
    int dstm[8];
#pragma unroll
    for (int v = 0; v < 8; ++v) dstm[v] = eidx[eb + v + half8];
#pragma unroll
    for (int nt = 0; nt < 8; ++nt) {
      int n = nt * 16 + r;
      float bia = Wn1_b[n];
#pragma unroll
      for (int v = 0; v < 8; ++v) {
        float val = mc[nt][v] + bia;
        if (act) val = fmaxf(val, 0.f);
        atomicAdd(&agg[(size_t)dstm[v] * HH + n], val);
      }
    }
  }
}

// ------------------------------------------------------------- node kernel ---
// x_new = act([x, agg, u[batch]] @ Wn2 + b); scatter into node_agg[batch].
// One wave per 32-node tile.
__global__ void __launch_bounds__(WPB * 32)
node_kernel(const unsigned* __restrict__ xbf, const float* __restrict__ aggf,
            const unsigned* __restrict__ ubf, const int* __restrict__ batch,
            const unsigned* __restrict__ Wn2T, const float* __restrict__ Wn2_b,
            float* __restrict__ x_out, float* __restrict__ node_agg, int act) {
  const int wave  = threadIdx.x >> 5;
  const int lane  = threadIdx.x & 31;
  const int r     = lane & 15;
  const int half8 = (lane & 16) >> 1;
  const int n0    = (blockIdx.x * WPB + wave) * 32;

  const unsigned* bx0  = xbf + (size_t)(n0 + r) * 64;
  const unsigned* bx1  = xbf + (size_t)(n0 + 16 + r) * 64;
  const float*    bag0 = aggf + (size_t)(n0 + r) * HH;
  const float*    bag1 = aggf + (size_t)(n0 + 16 + r) * HH;
  const unsigned* bu0  = ubf + (size_t)batch[n0 + r] * 64;
  const unsigned* bu1  = ubf + (size_t)batch[n0 + 16 + r] * 64;

  const v8f vzero = {0.f, 0.f, 0.f, 0.f, 0.f, 0.f, 0.f, 0.f};
  v8f acc0[8], acc1[8];
#pragma unroll
  for (int t = 0; t < 8; ++t) { acc0[t] = vzero; acc1[t] = vzero; }

  // K = 384, Wn2T rows are 192 dwords: x @ 0, agg @ 64, u @ 128
#pragma unroll
  for (int kb = 0; kb < 4; ++kb) {
    v16bf a0 = frag_bf16(bx0 + kb * 16), a1 = frag_bf16(bx1 + kb * 16);
#pragma unroll
    for (int nt = 0; nt < 8; ++nt) {
      v16bf b = frag_bf16(Wn2T + (size_t)(nt * 16 + r) * 192 + kb * 16);
      acc0[nt] = wmma_bf16(a0, b, acc0[nt]);
      acc1[nt] = wmma_bf16(a1, b, acc1[nt]);
    }
  }
#pragma unroll
  for (int kb = 0; kb < 4; ++kb) {
    v16bf a0 = frag_f32(bag0 + kb * 32), a1 = frag_f32(bag1 + kb * 32);
#pragma unroll
    for (int nt = 0; nt < 8; ++nt) {
      v16bf b = frag_bf16(Wn2T + (size_t)(nt * 16 + r) * 192 + 64 + kb * 16);
      acc0[nt] = wmma_bf16(a0, b, acc0[nt]);
      acc1[nt] = wmma_bf16(a1, b, acc1[nt]);
    }
  }
#pragma unroll
  for (int kb = 0; kb < 4; ++kb) {
    v16bf a0 = frag_bf16(bu0 + kb * 16), a1 = frag_bf16(bu1 + kb * 16);
#pragma unroll
    for (int nt = 0; nt < 8; ++nt) {
      v16bf b = frag_bf16(Wn2T + (size_t)(nt * 16 + r) * 192 + 128 + kb * 16);
      acc0[nt] = wmma_bf16(a0, b, acc0[nt]);
      acc1[nt] = wmma_bf16(a1, b, acc1[nt]);
    }
  }

#pragma unroll
  for (int t = 0; t < 2; ++t) {
    v8f* acc = t ? acc1 : acc0;
    const int nb = n0 + t * 16;
    int bm[8];
#pragma unroll
    for (int v = 0; v < 8; ++v) bm[v] = batch[nb + v + half8];
#pragma unroll
    for (int nt = 0; nt < 8; ++nt) {
      int n = nt * 16 + r;
      float bia = Wn2_b[n];
#pragma unroll
      for (int v = 0; v < 8; ++v) {
        int m = v + half8;
        float val = acc[nt][v] + bia;
        if (act) val = fmaxf(val, 0.f);
        x_out[(size_t)(nb + m) * HH + n] = val;
        atomicAdd(&node_agg[(size_t)bm[v] * HH + n], val);
      }
    }
  }
}

// ----------------------------------------------------------- global kernel ---
// u_new = act([node_agg, edge_agg, u] @ Wg + b); one wave per 32-graph tile.
__global__ void __launch_bounds__(WPB * 32)
glob_kernel(const float* __restrict__ node_agg, const float* __restrict__ edge_agg,
            const unsigned* __restrict__ ubf, const unsigned* __restrict__ WgT,
            const float* __restrict__ Wg_b, float* __restrict__ u_out, int act) {
  const int wave  = threadIdx.x >> 5;
  const int lane  = threadIdx.x & 31;
  const int r     = lane & 15;
  const int half8 = (lane & 16) >> 1;
  const int g0    = (blockIdx.x * WPB + wave) * 32;

  const float*    bn0 = node_agg + (size_t)(g0 + r) * HH;
  const float*    bn1 = node_agg + (size_t)(g0 + 16 + r) * HH;
  const float*    be0 = edge_agg + (size_t)(g0 + r) * HH;
  const float*    be1 = edge_agg + (size_t)(g0 + 16 + r) * HH;
  const unsigned* bu0 = ubf + (size_t)(g0 + r) * 64;
  const unsigned* bu1 = ubf + (size_t)(g0 + 16 + r) * 64;

  const v8f vzero = {0.f, 0.f, 0.f, 0.f, 0.f, 0.f, 0.f, 0.f};
  v8f acc0[8], acc1[8];
#pragma unroll
  for (int t = 0; t < 8; ++t) { acc0[t] = vzero; acc1[t] = vzero; }

#pragma unroll
  for (int kb = 0; kb < 4; ++kb) {
    v16bf a0 = frag_f32(bn0 + kb * 32), a1 = frag_f32(bn1 + kb * 32);
#pragma unroll
    for (int nt = 0; nt < 8; ++nt) {
      v16bf b = frag_bf16(WgT + (size_t)(nt * 16 + r) * 192 + kb * 16);
      acc0[nt] = wmma_bf16(a0, b, acc0[nt]);
      acc1[nt] = wmma_bf16(a1, b, acc1[nt]);
    }
  }
#pragma unroll
  for (int kb = 0; kb < 4; ++kb) {
    v16bf a0 = frag_f32(be0 + kb * 32), a1 = frag_f32(be1 + kb * 32);
#pragma unroll
    for (int nt = 0; nt < 8; ++nt) {
      v16bf b = frag_bf16(WgT + (size_t)(nt * 16 + r) * 192 + 64 + kb * 16);
      acc0[nt] = wmma_bf16(a0, b, acc0[nt]);
      acc1[nt] = wmma_bf16(a1, b, acc1[nt]);
    }
  }
#pragma unroll
  for (int kb = 0; kb < 4; ++kb) {
    v16bf a0 = frag_bf16(bu0 + kb * 16), a1 = frag_bf16(bu1 + kb * 16);
#pragma unroll
    for (int nt = 0; nt < 8; ++nt) {
      v16bf b = frag_bf16(WgT + (size_t)(nt * 16 + r) * 192 + 128 + kb * 16);
      acc0[nt] = wmma_bf16(a0, b, acc0[nt]);
      acc1[nt] = wmma_bf16(a1, b, acc1[nt]);
    }
  }

#pragma unroll
  for (int t = 0; t < 2; ++t) {
    v8f* acc = t ? acc1 : acc0;
    const int gb = g0 + t * 16;
#pragma unroll
    for (int nt = 0; nt < 8; ++nt) {
      int n = nt * 16 + r;
      float bia = Wg_b[n];
#pragma unroll
      for (int v = 0; v < 8; ++v) {
        float val = acc[nt][v] + bia;
        if (act) val = fmaxf(val, 0.f);
        u_out[(size_t)(gb + v + half8) * HH + n] = val;
      }
    }
  }
}

// ------------------------------------------------------------------ launch ---

extern "C" void kernel_launch(void* const* d_in, const int* in_sizes, int n_in,
                              void* d_out, int out_size, void* d_ws, size_t ws_size,
                              hipStream_t stream) {
  (void)in_sizes; (void)n_in; (void)out_size; (void)ws_size;

  const float* x_in  = (const float*)d_in[0];
  const float* ea_in = (const float*)d_in[1];
  const float* u_in  = (const float*)d_in[2];
  const int*   eidx  = (const int*)d_in[3];
  const int*   batch = (const int*)d_in[4];
  const float* We_W  = (const float*)d_in[5];
  const float* We_b  = (const float*)d_in[6];
  const float* Wn1_W = (const float*)d_in[7];
  const float* Wn1_b = (const float*)d_in[8];
  const float* Wn2_W = (const float*)d_in[9];
  const float* Wn2_b = (const float*)d_in[10];
  const float* Wg_W  = (const float*)d_in[11];
  const float* Wg_b  = (const float*)d_in[12];
  const float* bng_n = (const float*)d_in[13];
  const float* bnb_n = (const float*)d_in[14];
  const float* bng_e = (const float*)d_in[15];
  const float* bnb_e = (const float*)d_in[16];
  const float* bng_u = (const float*)d_in[17];
  const float* bnb_u = (const float*)d_in[18];

  float* out_x  = (float*)d_out;
  float* out_ea = out_x + (size_t)NN * HH;
  float* out_u  = out_ea + (size_t)EE * HH;

  // workspace carve (256B aligned)
  char* w = (char*)d_ws;
  auto carve = [&](size_t bytes) {
    void* p = (void*)w;
    w += (bytes + 255) & ~(size_t)255;
    return p;
  };
  unsigned*       xbf   = (unsigned*)carve((size_t)NN * HH * 2);
  unsigned*       eabf  = (unsigned*)carve((size_t)EE * HH * 2);
  unsigned*       ubf   = (unsigned*)carve((size_t)BB * HH * 2);
  float*          agg   = (float*)carve((size_t)NN * HH * 4);
  float*          nagg  = (float*)carve((size_t)BB * HH * 4);
  float*          eagg  = (float*)carve((size_t)BB * HH * 4);
  float*          stats = (float*)carve(768 * 4);          // [x | ea | u] sums+sumsq
  unsigned short* WeT   = (unsigned short*)carve((size_t)LLAYERS * 128 * 512 * 2);
  unsigned short* Wn1T  = (unsigned short*)carve((size_t)LLAYERS * 128 * 256 * 2);
  unsigned short* Wn2T  = (unsigned short*)carve((size_t)LLAYERS * 128 * 384 * 2);
  unsigned short* WgT   = (unsigned short*)carve((size_t)LLAYERS * 128 * 384 * 2);

  // --- once: transpose + convert all weights to bf16 [H][K] ---
  for (int l = 0; l < LLAYERS; ++l) {
    wt_cvt_kernel<<<(128 * 512 + 255) / 256, 256, 0, stream>>>(
        We_W + (size_t)l * 512 * 128, WeT + (size_t)l * 128 * 512, 512);
    wt_cvt_kernel<<<(128 * 256 + 255) / 256, 256, 0, stream>>>(
        Wn1_W + (size_t)l * 256 * 128, Wn1T + (size_t)l * 128 * 256, 256);
    wt_cvt_kernel<<<(128 * 384 + 255) / 256, 256, 0, stream>>>(
        Wn2_W + (size_t)l * 384 * 128, Wn2T + (size_t)l * 128 * 384, 384);
    wt_cvt_kernel<<<(128 * 384 + 255) / 256, 256, 0, stream>>>(
        Wg_W + (size_t)l * 384 * 128, WgT + (size_t)l * 128 * 384, 384);
  }

  const int nN = NN * HH, nB = BB * HH;

  for (int i = 0; i < LLAYERS; ++i) {
    const int act = (i < LLAYERS - 1) ? 1 : 0;

    // zero accumulators + BN stats
    zero_kernel<<<(nN + 255) / 256, 256, 0, stream>>>(agg, nN);
    zero_kernel<<<(nB + 255) / 256, 256, 0, stream>>>(nagg, nB);
    zero_kernel<<<(nB + 255) / 256, 256, 0, stream>>>(eagg, nB);
    zero_kernel<<<3, 256, 0, stream>>>(stats, 768);

    // bf16 mirrors: layer 0 from inputs; later layers come fused out of BN
    if (i == 0) {
      cvt_bf16_kernel<<<(NN * HH / 2 + 255) / 256, 256, 0, stream>>>(x_in, xbf, NN * HH / 2);
      cvt_bf16_kernel<<<(EE * HH / 2 + 255) / 256, 256, 0, stream>>>(ea_in, eabf, EE * HH / 2);
      cvt_bf16_kernel<<<(BB * HH / 2 + 255) / 256, 256, 0, stream>>>(u_in, ubf, BB * HH / 2);
    }

    // fused edge MLP + message MLP + scatter sums
    edge_kernel<<<EE / (32 * WPB), WPB * 32, 0, stream>>>(
        xbf, eabf, ubf, eidx, batch,
        (const unsigned*)(WeT + (size_t)i * 128 * 512), We_b + i * HH,
        (const unsigned*)(Wn1T + (size_t)i * 128 * 256), Wn1_b + i * HH,
        out_ea, agg, eagg, act);

    // node update + node_agg scatter
    node_kernel<<<NN / (32 * WPB), WPB * 32, 0, stream>>>(
        xbf, agg, ubf, batch,
        (const unsigned*)(Wn2T + (size_t)i * 128 * 384), Wn2_b + i * HH,
        out_x, nagg, act);

    // global update
    glob_kernel<<<BB / (32 * WPB), WPB * 32, 0, stream>>>(
        nagg, eagg, ubf,
        (const unsigned*)(WgT + (size_t)i * 128 * 384), Wg_b + i * HH,
        out_u, act);

    // BatchNorm (layers 0,1): normalize in place + emit next layer's bf16 mirror
    if (i < LLAYERS - 1) {
      bn_stat_kernel<<<512, 256, 0, stream>>>(out_x, stats, NN);
      bn_norm_kernel<<<(NN * HH / 2 + 255) / 256, 256, 0, stream>>>(
          out_x, xbf, stats, bng_n + i * HH, bnb_n + i * HH, NN);
      bn_stat_kernel<<<512, 256, 0, stream>>>(out_ea, stats + 256, EE);
      bn_norm_kernel<<<(EE * HH / 2 + 255) / 256, 256, 0, stream>>>(
          out_ea, eabf, stats + 256, bng_e + i * HH, bnb_e + i * HH, EE);
      bn_stat_kernel<<<64, 256, 0, stream>>>(out_u, stats + 512, BB);
      bn_norm_kernel<<<(BB * HH / 2 + 255) / 256, 256, 0, stream>>>(
          out_u, ubf, stats + 512, bng_u + i * HH, bnb_u + i * HH, BB);
    }
  }
}
